// DenseConcatBlocks_56607668961287
// MI455X (gfx1250) — matrix operand order repaired and checked by reading, not compiled, not run on hardware
//
#include <hip/hip_runtime.h>
#include <hip/hip_bf16.h>

// DenseConcatBlocks on MI455X (gfx1250): bf16 implicit-GEMM conv via
// v_wmma_f32_16x16x32_bf16. M=16 hidden channels, N=16 pixels/row-tile.
// K indexing: k = 16*(c*9+ky) + kx, kx in 0..15 (kx>=9 zero-weighted), so a
// K=32 chunk is exactly two kernel rows (one per lane half) and each B
// fragment is a contiguous 32-byte LDS strip -> 2x ds_load_b128, no movs.
// Halo staging uses GLOBAL_LOAD_ASYNC_TO_LDS_B128 (ASYNCcnt) when available.
// Workgroup = 16x32 pixel tile, 8 wave32; each wave owns 4 rows (4 accums).

#define NUM_BLOCKS 50
#define HIDDEN 16
#define HH 128
#define WW 128
#define CMAX 51   // 1 + NUM_BLOCKS feature channels
#define SROWS 40  // staged halo rows per channel (32 outputs + 8 halo)
#define SCOLS 32  // staged cols per row (window x0-8 .. x0+23)

#if defined(__AMDGCN__) && __has_builtin(__builtin_amdgcn_global_load_async_to_lds_b128)
#define HAVE_ASYNC 1
#else
#define HAVE_ASYNC 0
#endif

typedef __attribute__((ext_vector_type(16))) __bf16 v16bf;
typedef __attribute__((ext_vector_type(8)))  float  v8f;

#if HAVE_ASYNC
typedef int v4i_gcc __attribute__((vector_size(16)));
typedef __attribute__((address_space(1))) v4i_gcc* gvec_p;   // global int4*
typedef __attribute__((address_space(3))) v4i_gcc* lvec_p;   // LDS int4*
#endif

union FragU { v16bf v; uint4 q[2]; };

// ---------------------------------------------------------------------------
// Convert input x (f32, [16,1,128,128]) into feats channel 0 (bf16).
// ---------------------------------------------------------------------------
__global__ __launch_bounds__(256) void init_feats_kernel(
    const float* __restrict__ x, __bf16* __restrict__ feats)
{
    int tid = blockIdx.x * 256 + threadIdx.x;      // 16*128*128 threads
    int b = tid >> 14;
    int p = tid & 16383;
    feats[(size_t)b * CMAX * (HH * WW) + p] = (__bf16)x[tid];
}

// ---------------------------------------------------------------------------
// Pack block-i weights w1[i, M, c, ky, kx] into the exact WMMA A-fragment
// lane layout (16x32 bf16 per K-chunk of 32).
// K index: k = 16*(c*9+ky) + kx, kx in 0..15, kx>=9 (and rows past 9C) zero.
// Output: apack[(chunk*32 + lane)*16 + j]  (one v16bf per lane per chunk).
// ---------------------------------------------------------------------------
__global__ __launch_bounds__(256) void pack_weights_kernel(
    const float* __restrict__ w1, __bf16* __restrict__ apack,
    int blk, int C, int nchunks)
{
    int tid = blockIdx.x * 256 + threadIdx.x;
    int total = nchunks * 512;
    if (tid >= total) return;

    int j    = tid & 15;         // half index within lane's v16bf
    int lane = (tid >> 4) & 31;
    int ch   = tid >> 9;         // K-chunk

    // A-matrix 16x32 bf16 layout (contiguous K, matching B fragments):
    //   lanes 0-15 : M=lane,    element j holds K = j
    //   lanes 16-31: M=lane-16, element j holds K = 16+j
    int K = ((lane & 16) ? 16 : 0) + j;
    int M = lane & 15;

    int k   = ch * 32 + K;
    int rho = k >> 4;            // kernel-row index = c*9 + ky
    int kx  = k & 15;
    int c   = rho / 9;
    int ky  = rho - 9 * c;
    float val = 0.0f;
    if (kx < 9 && c < C)
        val = w1[((((size_t)blk * HIDDEN + M) * CMAX + c) * 9 + ky) * 9 + kx];
    apack[tid] = (__bf16)val;    // tid == (ch*32+lane)*16 + j
}

// ---------------------------------------------------------------------------
// One dense block: conv(9x9, C->16) + relu + bias, then per-pixel weighted
// channel sum -> sigmoid -> write feats channel (1+blk) as bf16 and the fp32
// output plane.
// Grid: (8 x-tiles, 4 y-tiles, 16 batch), 256 threads (8 wave32).
// Workgroup: 16x32 output pixels; each wave: 4 rows of 16 pixels.
// ---------------------------------------------------------------------------
__global__ __launch_bounds__(256) void dense_block_kernel(
    const __bf16* __restrict__ feats,
    const v16bf*  __restrict__ apack,
    const float*  __restrict__ b1,
    const float*  __restrict__ w2,
    const float*  __restrict__ b2,
    float*        __restrict__ out,
    __bf16*       __restrict__ featsOut,
    int blk, int C, int nchunks)
{
    extern __shared__ __bf16 slab[];       // [C][SROWS rows][SCOLS cols]
    const int tid  = threadIdx.x;
    const int lane = tid & 31;
    const int wave = tid >> 5;
    const int x0 = blockIdx.x * 16;
    const int y0 = blockIdx.y * 32;
    const int b  = blockIdx.z;

    // ---- stage halo slab: rows y0-4 .. y0+35, cols x0-8 .. x0+23 ----------
    // (window shifted -8 so interior global addresses are 16B aligned)
    const int nrows = C * SROWS;
    const int xs = x0 - 8;
    for (int idx = tid; idx < nrows; idx += 256) {
        int c  = idx / SROWS;
        int rr = idx - c * SROWS;
        int yy = y0 + rr - 4;
        __bf16* dst = &slab[idx * SCOLS];
        if ((unsigned)yy < (unsigned)HH) {
            const __bf16* src = feats + (((size_t)b * CMAX + c) * HH + yy) * WW;
            if (xs >= 0 && xs + (SCOLS - 1) < WW) {
#if HAVE_ASYNC
                // async global->LDS DMA, 64B per row as 4x b128 (ASYNCcnt)
                gvec_p g = (gvec_p)(src + xs);
                lvec_p l = (lvec_p)dst;
                __builtin_amdgcn_global_load_async_to_lds_b128(g, l, 0,  0);
                __builtin_amdgcn_global_load_async_to_lds_b128(g, l, 16, 0);
                __builtin_amdgcn_global_load_async_to_lds_b128(g, l, 32, 0);
                __builtin_amdgcn_global_load_async_to_lds_b128(g, l, 48, 0);
#else
                const uint2* s2 = (const uint2*)(src + xs);   // 16B aligned
                uint2* d2 = (uint2*)dst;
                #pragma unroll
                for (int q = 0; q < SCOLS / 4; ++q) d2[q] = s2[q];
#endif
            } else {
                for (int colp = 0; colp < SCOLS; ++colp) {
                    int xx = xs + colp;
                    dst[colp] = ((unsigned)xx < (unsigned)WW) ? src[xx]
                                                              : (__bf16)0.0f;
                }
            }
        } else {
            uint2 z; z.x = 0u; z.y = 0u;
            uint2* d2 = (uint2*)dst;
            #pragma unroll
            for (int q = 0; q < SCOLS / 4; ++q) d2[q] = z;
        }
    }
#if HAVE_ASYNC
#if __has_builtin(__builtin_amdgcn_s_wait_asynccnt)
    __builtin_amdgcn_s_wait_asynccnt(0);
#else
    asm volatile("s_wait_asynccnt 0" ::: "memory");
#endif
#endif
    __syncthreads();

    // ---- K loop: one chunk = two kernel rows (one per lane half) ----------
    // B fragment for row-tile rr2 = 32 contiguous bytes at
    //   slab_bytes + addr + rr2*SCOLS*2;  strip base col = n+4 (kx tap 0 at
    //   col n+4 <-> image x = x0+n-4).  Junk taps (kx>=9) may overrun into
    //   the next slab row / LDS OOB (returns 0) - all zero-weighted in A.
    const int r0 = wave * 4;            // this wave's four output rows
    const int n  = lane & 15;           // B/N column owned by this lane
    const int hi = (lane >> 4) & 1;     // lane half -> row parity within chunk

    const char* slabB = (const char*)slab;
    int ky   = hi;                           // kernel row within channel
    int addr = (r0 + ky) * (2 * SCOLS) + 2 * (n + 4);
    const v16bf* ap = apack + lane;

    v8f acc0 = {}, acc1 = {}, acc2 = {}, acc3 = {};
    for (int t = 0; t < nchunks; ++t) {
        v16bf a = ap[t * 32];               // 32B per-lane A fragment

        FragU f0, f1, f2, f3;               // rows r0 .. r0+3
        {
            const uint4* p0 = (const uint4*)(slabB + addr);
            f0.q[0] = p0[0];  f0.q[1] = p0[1];
            const uint4* p1 = (const uint4*)(slabB + addr + 2 * SCOLS);
            f1.q[0] = p1[0];  f1.q[1] = p1[1];
            const uint4* p2 = (const uint4*)(slabB + addr + 4 * SCOLS);
            f2.q[0] = p2[0];  f2.q[1] = p2[1];
            const uint4* p3 = (const uint4*)(slabB + addr + 6 * SCOLS);
            f3.q[0] = p3[0];  f3.q[1] = p3[1];
        }
        acc0 = __builtin_amdgcn_wmma_f32_16x16x32_bf16(
                   false, a, false, f0.v, (short)0, acc0, false, false);
        acc1 = __builtin_amdgcn_wmma_f32_16x16x32_bf16(
                   false, a, false, f1.v, (short)0, acc1, false, false);
        acc2 = __builtin_amdgcn_wmma_f32_16x16x32_bf16(
                   false, a, false, f2.v, (short)0, acc2, false, false);
        acc3 = __builtin_amdgcn_wmma_f32_16x16x32_bf16(
                   false, a, false, f3.v, (short)0, acc3, false, false);

        // advance two kernel rows; hop to next channel slab when ky wraps
        ky += 2;
        int cross = (ky >= 9);
        addr += cross ? ((SROWS - 7) * (2 * SCOLS)) : (2 * (2 * SCOLS));
        ky   -= cross ? 9 : 0;
    }

    // ---- epilogue: relu+bias, weighted channel sum, sigmoid --------------
    const int mbase = hi ? 8 : 0;
    float b1v[8], w2v[8];
    #pragma unroll
    for (int j = 0; j < 8; ++j) {
        b1v[j] = b1[blk * HIDDEN + mbase + j];
        w2v[j] = w2[blk * HIDDEN + mbase + j];
    }
    const float b2s = b2[blk];

    v8f accs[4] = {acc0, acc1, acc2, acc3};
    #pragma unroll
    for (int rr2 = 0; rr2 < 4; ++rr2) {
        v8f acc = accs[rr2];
        float part = 0.0f;
        #pragma unroll
        for (int j = 0; j < 8; ++j) {
            float h = acc[j] + b1v[j];
            h = h > 0.0f ? h : 0.0f;
            part += h * w2v[j];
        }
        part += __shfl_xor(part, 16, 32);        // combine M 0-7 with 8-15
        float s = 1.0f / (1.0f + __expf(-(part + b2s)));
        if (lane < 16) {
            int y = y0 + r0 + rr2;
            int x = x0 + n;
            out[(((size_t)b * NUM_BLOCKS + blk) * HH + y) * WW + x] = s;
            featsOut[(((size_t)b * CMAX + (1 + blk)) * HH + y) * WW + x] =
                (__bf16)s;
        }
    }
}

// ---------------------------------------------------------------------------
extern "C" void kernel_launch(void* const* d_in, const int* in_sizes, int n_in,
                              void* d_out, int out_size, void* d_ws, size_t ws_size,
                              hipStream_t stream)
{
    (void)in_sizes; (void)n_in; (void)out_size; (void)ws_size;
    const float* x  = (const float*)d_in[0];
    const float* w1 = (const float*)d_in[1];
    const float* b1 = (const float*)d_in[2];
    const float* w2 = (const float*)d_in[3];
    const float* b2 = (const float*)d_in[4];
    float* out = (float*)d_out;

    // ws layout: [feats bf16: 16*51*128*128] [apack bf16: Σ nchunks * 512]
    __bf16* feats = (__bf16*)d_ws;
    const size_t featsElems = (size_t)16 * CMAX * HH * WW;   // 13.37M (26.7MB)
    __bf16* apackAll = feats + featsElems;                   // 32B-aligned

    init_feats_kernel<<<(16 * HH * WW) / 256, 256, 0, stream>>>(x, feats);

    size_t chunkBase = 0;
    for (int i = 0; i < NUM_BLOCKS; ++i) {
        int C = 1 + i;
        int nchunks = (9 * C + 1) / 2;       // two kernel rows per chunk
        __bf16* ap = apackAll + chunkBase * 512;

        pack_weights_kernel<<<nchunks * 2, 256, 0, stream>>>(w1, ap, i, C,
                                                             nchunks);

        size_t smem = (size_t)C * SROWS * SCOLS * sizeof(__bf16); // <=128 KB
        dense_block_kernel<<<dim3(8, 4, 16), 256, smem, stream>>>(
            feats, (const v16bf*)ap, b1, w2, b2, out, feats, i, C, nchunks);

        chunkBase += nchunks;
    }
}
